// StackedGRUDClassifier_12378095747095
// MI455X (gfx1250) — matrix operand order repaired and checked by reading, not compiled, not run on hardware
//
#include <hip/hip_runtime.h>
#include <math.h>

// ---------------------------------------------------------------------------
// GRU-D stacked recurrent net, persistent-kernel WMMA implementation (gfx1250)
// B=256, D=20, T=1024, H=120.  One workgroup = 16 batch rows, 8 waves.
// Every matmul is v_wmma_f32_16x16x32_f16: M=16 (batch tile), N tile = 16
// (wave w owns hidden cols [16w,16w+16)), K chunked by 32.
//
// Round-1 fixes: recurrent f32 state (h1,h2,h_d) is register-resident per
// wave; all LDS A-buffer segments are padded so epilogue stores need no
// lane-divergent guards (pad cols compute exact zeros with zero-padded
// weights); r*h_d goes to a second A-buffer to remove WAR barriers.
// ---------------------------------------------------------------------------

typedef __attribute__((ext_vector_type(16))) _Float16 v16h;
typedef __attribute__((ext_vector_type(8)))  _Float16 v8h;
typedef __attribute__((ext_vector_type(8)))  float    v8f;

#define B_    256
#define D_    20
#define T_    1024
#define H_    120
#define H3_   40
#define H9_   13
#define H27_  4
#define OUT_  6
#define AUX_  2
#define HP_   128          // H padded to N-tiles

// Padded A-buffer layouts (f16, row-major, 16 rows):
//  A1/A1t: [xi 0..31 | h1d 32..159 | m 160..179 | 0 180..191]  stride 192, 6 K-chunks
//  A2/A2t: [h1n 0..127 | h2d 128..255 | m 256..275 | 0 276..287] stride 288, 9 K-chunks
#define A1S   192
#define A1HD  32
#define A1M   160
#define A2S   288
#define A2H2  128
#define A2M   256

// Packed-weight regions (halves). Region = KT*NT*512,
// layout: ((kt*NT + nt)*32 + lane)*16 + j ;  n = nt*16+(lane&15),
// packed k = kt*32+(lane>>4)*8+(j&7)+(j>>3)*16, remapped to source K (zero if unmapped).
static const long OFF_GX  = 0;        // KT=1 NT=2  -> 1024
static const long OFF_GH1 = 1024;     // KT=1 NT=8  -> 4096
static const long OFF_GH2 = 5120;     // 4096
static const long OFF_Z1  = 9216;     // KT=6 NT=8  -> 24576
static const long OFF_R1  = 33792;
static const long OFF_T1  = 58368;
static const long OFF_Z2  = 82944;    // KT=9 NT=8  -> 36864
static const long OFF_R2  = 119808;
static const long OFF_T2  = 156672;
static const long OFF_F1  = 193536;   // KT=4 NT=3  -> 6144
static const long OFF_A1W = 199680;   // 6144
static const long PW_TOTAL= 205824;   // halves (~412 KB workspace)

// ---------------------------------------------------------------------------

__device__ __forceinline__ v8f wmma32(v16h a, v16h b, v8f c) {
  return __builtin_amdgcn_wmma_f32_16x16x32_f16(false, a, false, b,
                                                (short)0, c, false, false);
}

// A-fragment (16x32 f16) per ISA layout: lanes 0-15 row M=lane hold
// K=k0+0..7 (v0..7) and k0+16..23 (v8..15); lanes 16-31 hold k0+8..15, k0+24..31.
__device__ __forceinline__ v16h load_af(const _Float16* A, int lda, int k0,
                                        int lane) {
  int row = lane & 15;
  int hs  = (lane >> 4) & 1;
  const _Float16* p = A + row * lda + k0 + hs * 8;
  v8h lo = *(const v8h*)(p);
  v8h hi = *(const v8h*)(p + 16);
  v16h a;
#pragma unroll
  for (int i = 0; i < 8; ++i) { a[i] = lo[i]; a[i + 8] = hi[i]; }
  return a;
}

// Pre-packed B fragment: contiguous 32 bytes per lane.
__device__ __forceinline__ v16h load_bp(const _Float16* pw, long base, int NT,
                                        int kt, int nt, int lane) {
  return *(const v16h*)(pw + base + (((long)kt * NT + nt) * 32 + lane) * 16);
}

__device__ __forceinline__ float sigmoidf_(float x) {
  return 1.f / (1.f + __expf(-x));
}
__device__ __forceinline__ float lrelu_(float x) {
  return x > 0.f ? x : 0.01f * x;
}

// ---------------------------------------------------------------------------
// Prep kernel: pack f32 row-major W[N][K] into WMMA-B f16 fragments with a
// per-matrix packed-K -> source-K remap (matches padded A layouts).
// ---------------------------------------------------------------------------
__device__ __forceinline__ int remap_k(int kp, int K, int layout) {
  if (layout == 1) {                 // layer-1 gates: src [xi20|h1d120|m20]
    if (kp < 32)  return (kp < D_) ? kp : -1;
    if (kp < 160) { int q = kp - 32;  return (q < H_) ? D_ + q : -1; }
    if (kp < 180) return (D_ + H_) + (kp - 160);
    return -1;
  }
  if (layout == 2) {                 // layer-2 gates: src [h1n120|h2d120|m20]
    if (kp < 128) return (kp < H_) ? kp : -1;
    if (kp < 256) { int q = kp - 128; return (q < H_) ? H_ + q : -1; }
    if (kp < 276) return 2 * H_ + (kp - 256);
    return -1;
  }
  return (kp < K) ? kp : -1;         // identity with bound check
}

__device__ void pack_region(_Float16* pw, long base, const float* W, int N,
                            int K, int NT, int KT, int layout, int tid,
                            int stride) {
  int total = KT * NT * 512;
  for (int idx = tid; idx < total; idx += stride) {
    int kt = idx / (NT * 512);
    int r  = idx - kt * NT * 512;
    int nt = r >> 9;
    int r2 = r & 511;
    int l  = r2 >> 4;
    int j  = r2 & 15;
    int n  = nt * 16 + (l & 15);
    int kp = kt * 32 + (l >> 4) * 8 + (j & 7) + (j >> 3) * 16;
    int k  = remap_k(kp, K, layout);
    float v = (n < N && k >= 0) ? W[(long)n * K + k] : 0.f;
    pw[base + idx] = (_Float16)v;
  }
}

__global__ void __launch_bounds__(256) pack_weights_kernel(
    const float* Wgx, const float* Wgh1, const float* Wr1, const float* Wz1,
    const float* Wt1, const float* Wgh2, const float* Wr2, const float* Wz2,
    const float* Wt2, const float* Wf1, const float* Wa1, _Float16* pw) {
  int tid    = blockIdx.x * blockDim.x + threadIdx.x;
  int stride = gridDim.x * blockDim.x;
  pack_region(pw, OFF_GX,  Wgx,  D_,  D_,      2, 1, 0, tid, stride);
  pack_region(pw, OFF_GH1, Wgh1, H_,  D_,      8, 1, 0, tid, stride);
  pack_region(pw, OFF_GH2, Wgh2, H_,  D_,      8, 1, 0, tid, stride);
  pack_region(pw, OFF_Z1,  Wz1,  H_,  160,     8, 6, 1, tid, stride);
  pack_region(pw, OFF_R1,  Wr1,  H_,  160,     8, 6, 1, tid, stride);
  pack_region(pw, OFF_T1,  Wt1,  H_,  160,     8, 6, 1, tid, stride);
  pack_region(pw, OFF_Z2,  Wz2,  H_,  260,     8, 9, 2, tid, stride);
  pack_region(pw, OFF_R2,  Wr2,  H_,  260,     8, 9, 2, tid, stride);
  pack_region(pw, OFF_T2,  Wt2,  H_,  260,     8, 9, 2, tid, stride);
  pack_region(pw, OFF_F1,  Wf1,  H3_, H_,      3, 4, 0, tid, stride);
  pack_region(pw, OFF_A1W, Wa1,  H3_, H_,      3, 4, 0, tid, stride);
}

// ---------------------------------------------------------------------------
// Main persistent kernel.
// ---------------------------------------------------------------------------
struct Params {
  const float* inputs;   // (B,4,T,D)
  const float* x_mean;
  const float* b_gx;  const float* b_gh1;
  const float* b_r1;  const float* b_z1;  const float* b_t1;
  const float* b_gh2; const float* b_r2;  const float* b_z2;  const float* b_t2;
  const float* b_f1;  const float* W_f2;  const float* b_f2;
  const float* W_f3;  const float* b_f3;  const float* W_f4;  const float* b_f4;
  const float* b_a1;  const float* W_a2;  const float* b_a2;
  const float* W_a3;  const float* b_a3;  const float* W_a4;  const float* b_a4;
  const _Float16* pw;
  float* out;
};

__global__ void __launch_bounds__(256) grud_main(Params p) {
  __shared__ __align__(32) _Float16 sA1 [16 * A1S];   // [xi | h1d     | m | 0]
  __shared__ __align__(32) _Float16 sA1t[16 * A1S];   // [xi | r1*h1d  | m | 0]
  __shared__ __align__(32) _Float16 sA2 [16 * A2S];   // [h1n | h2d    | m | 0]
  __shared__ __align__(32) _Float16 sA2t[16 * A2S];   // [h1n | r2*h2d | m | 0]
  __shared__ __align__(32) _Float16 sAd [16 * 32];    // delta (K pad 32)
  __shared__ __align__(32) _Float16 sAh [16 * HP_];   // h2n (fc input)
  __shared__ float sX[16 * 32], sM[16 * 32], sXp[16 * 32];
  __shared__ float sFC1[16 * 48], sFC2[16 * 16], sFC3[16 * 8];
  __shared__ float sB[1768];

  float* sBgx  = sB;          float* sXme  = sB + 32;
  float* sBgh1 = sB + 64;
  float* sBz1  = sB + 192;    float* sBr1  = sB + 320;   float* sBt1 = sB + 448;
  float* sBgh2 = sB + 576;
  float* sBz2  = sB + 704;    float* sBr2  = sB + 832;   float* sBt2 = sB + 960;
  float* sBf1  = sB + 1088;   // 48
  float* sWf2  = sB + 1136;   // 520
  float* sBf2  = sB + 1656;   // 16
  float* sWf3  = sB + 1672;   // 56
  float* sBf3  = sB + 1728;   // 8
  float* sWf4  = sB + 1736;   // 24
  float* sBf4  = sB + 1760;   // 8

  const int tid  = threadIdx.x;
  const int lane = tid & 31;
  const int wv   = tid >> 5;          // wave w owns hidden cols [16w,16w+16)
  const int r0   = blockIdx.x * 16;   // 16 batch rows per block
  const int nn   = lane & 15;
  const int hs   = lane >> 4;
  const int col  = wv * 16 + nn;

  const _Float16* pw = p.pw;
  float* out_main = p.out;                               // (B, OUT)
  float* out_tr   = p.out + (long)B_ * OUT_;             // (B, T, OUT)
  float* out_aux  = out_tr + (long)B_ * T_ * OUT_;       // (B, AUX)
  float* out_h2   = out_aux + (long)B_ * AUX_;           // (B, H)

  // Register-resident recurrent state for this wave's 8 (row,col) cells.
  float h1v[8], h2v[8];
#pragma unroll
  for (int j = 0; j < 8; ++j) { h1v[j] = 0.f; h2v[j] = 0.f; }

  // ---- init: zero pad regions (never rewritten), stage biases ------------
  for (int i = tid; i < 16 * 12; i += 256) {           // x/m/xp/delta col pads
    int r = i / 12, k = i % 12;
    sX[r * 32 + D_ + k] = 0.f; sM[r * 32 + D_ + k] = 0.f;
    sXp[r * 32 + D_ + k] = 0.f;
    sAd[r * 32 + D_ + k] = (_Float16)0.f;
  }
  for (int i = tid; i < 16 * 12; i += 256) {           // A1 K-pad 180..191
    int r = i / 12, k = i % 12;
    sA1 [r * A1S + 180 + k] = (_Float16)0.f;
    sA1t[r * A1S + 180 + k] = (_Float16)0.f;
  }
  for (int i = tid; i < 16 * 12; i += 256) {           // A2 K-pad 276..287
    int r = i / 12, k = i % 12;
    sA2 [r * A2S + 276 + k] = (_Float16)0.f;
    sA2t[r * A2S + 276 + k] = (_Float16)0.f;
  }
  for (int i = tid; i < 32; i += 256) {
    sBgx[i] = (i < D_) ? p.b_gx[i] : 0.f;
    sXme[i] = (i < D_) ? p.x_mean[i] : 0.f;
  }
  for (int i = tid; i < 128; i += 256) {
    bool ok = (i < H_);
    sBgh1[i] = ok ? p.b_gh1[i] : 0.f;  sBz1[i] = ok ? p.b_z1[i] : 0.f;
    sBr1[i]  = ok ? p.b_r1[i]  : 0.f;  sBt1[i] = ok ? p.b_t1[i] : 0.f;
    sBgh2[i] = ok ? p.b_gh2[i] : 0.f;  sBz2[i] = ok ? p.b_z2[i] : 0.f;
    sBr2[i]  = ok ? p.b_r2[i]  : 0.f;  sBt2[i] = ok ? p.b_t2[i] : 0.f;
  }
  for (int i = tid; i < 48;  i += 256) sBf1[i] = (i < H3_) ? p.b_f1[i] : 0.f;
  for (int i = tid; i < 520; i += 256) sWf2[i] = p.W_f2[i];
  for (int i = tid; i < 13;  i += 256) sBf2[i] = p.b_f2[i];
  for (int i = tid; i < 52;  i += 256) sWf3[i] = p.W_f3[i];
  for (int i = tid; i < 4;   i += 256) sBf3[i] = p.b_f3[i];
  for (int i = tid; i < 24;  i += 256) sWf4[i] = p.W_f4[i];
  for (int i = tid; i < 6;   i += 256) sBf4[i] = p.b_f4[i];
  __syncthreads();

  // per-wave column biases (register-resident, loaded once before the scan)
  const float bgh1c = sBgh1[col], bz1c = sBz1[col], br1c = sBr1[col];
  const float bt1c  = sBt1[col],  bgh2c = sBgh2[col], bz2c = sBz2[col];
  const float br2c  = sBr2[col],  bt2c = sBt2[col];
  const float bgxc  = sBgx[col & 31], xmec = sXme[col & 31];
  const float bf1c  = (wv < 3) ? sBf1[col] : 0.f;

  // -------------------------- scan over time ------------------------------
  for (int t = 0; t < T_; ++t) {
    // 1. stage inputs: 16 rows x 4 channels x 5 float4 = 320 vector loads
    for (int i = tid; i < 320; i += 256) {
      int rc = i / 5, v = i - rc * 5;
      int r = rc >> 2, c = rc & 3;
      const float4 val = *(const float4*)(
          p.inputs + (((long)(r0 + r) * 4 + c) * T_ + t) * D_ + v * 4);
      float e[4] = {val.x, val.y, val.z, val.w};
      int d0 = v * 4;
      if (c == 0) {
#pragma unroll
        for (int q = 0; q < 4; ++q) sX[r * 32 + d0 + q] = e[q];
      } else if (c == 1) {
#pragma unroll
        for (int q = 0; q < 4; ++q) {
          int d = d0 + q;
          sM[r * 32 + d] = e[q];
          _Float16 mh = (_Float16)e[q];
          sA1 [r * A1S + A1M + d] = mh;  sA1t[r * A1S + A1M + d] = mh;
          sA2 [r * A2S + A2M + d] = mh;  sA2t[r * A2S + A2M + d] = mh;
        }
      } else if (c == 2) {
#pragma unroll
        for (int q = 0; q < 4; ++q) sAd[r * 32 + d0 + q] = (_Float16)e[q];
      } else {
#pragma unroll
        for (int q = 0; q < 4; ++q) sXp[r * 32 + d0 + q] = e[q];
      }
    }
    __syncthreads();

    // warm near caches for this wave's layer-2 fragments
#pragma unroll
    for (int kt = 0; kt < 9; ++kt) {
      long o = (((long)kt * 8 + wv) * 32 + lane) * 16;
      __builtin_prefetch(pw + OFF_Z2 + o, 0, 3);
      __builtin_prefetch(pw + OFF_R2 + o, 0, 3);
      __builtin_prefetch(pw + OFF_T2 + o, 0, 3);
    }

    // 2. gh1 (all waves) and gx/xi (waves 0-1): A = delta (16x32)
    float hdv[8];
    {
      v16h ad = load_af(sAd, 32, 0, lane);
      v8f acc;
#pragma unroll
      for (int j = 0; j < 8; ++j) acc[j] = bgh1c;
      acc = wmma32(ad, load_bp(pw, OFF_GH1, 8, 0, wv, lane), acc);
#pragma unroll
      for (int j = 0; j < 8; ++j) {
        int row  = j + hs * 8;
        float g  = __expf(-fmaxf(acc[j], 0.f));
        float hd = g * h1v[j];
        hdv[j] = hd;
        sA1[row * A1S + A1HD + col] = (_Float16)hd;   // pad cols write 0
      }
      if (wv < 2) {
        v8f ax;
#pragma unroll
        for (int j = 0; j < 8; ++j) ax[j] = bgxc;
        ax = wmma32(ad, load_bp(pw, OFF_GX, 2, 0, wv, lane), ax);
#pragma unroll
        for (int j = 0; j < 8; ++j) {
          int row  = j + hs * 8;
          float g  = __expf(-fmaxf(ax[j], 0.f));
          float xv = sX[row * 32 + col], mv = sM[row * 32 + col];
          float xi = mv * xv + (1.f - mv) * (g * sXp[row * 32 + col] +
                                             (1.f - g) * xmec);
          _Float16 xh = (_Float16)xi;                  // pad cols compute 0
          sA1 [row * A1S + col] = xh;
          sA1t[row * A1S + col] = xh;
        }
      }
    }
    __syncthreads();

    // 3. z1, r1 over A1 (K = 192, 6 chunks); write r1*h1d into A1t
    float z1v[8];
    {
      v8f az, ar;
#pragma unroll
      for (int j = 0; j < 8; ++j) { az[j] = bz1c; ar[j] = br1c; }
#pragma unroll
      for (int kt = 0; kt < 6; ++kt) {
        v16h a = load_af(sA1, A1S, kt * 32, lane);
        az = wmma32(a, load_bp(pw, OFF_Z1, 8, kt, wv, lane), az);
        ar = wmma32(a, load_bp(pw, OFF_R1, 8, kt, wv, lane), ar);
      }
#pragma unroll
      for (int j = 0; j < 8; ++j) {
        int row = j + hs * 8;
        z1v[j] = sigmoidf_(az[j]);
        sA1t[row * A1S + A1HD + col] =
            (_Float16)(sigmoidf_(ar[j]) * hdv[j]);
      }
    }
    __syncthreads();

    // 4. t1 over A1t -> h1n (regs + A2); gh2 -> h2d (regs + A2). Disjoint.
    float hd2v[8];
    {
      v8f at;
#pragma unroll
      for (int j = 0; j < 8; ++j) at[j] = bt1c;
#pragma unroll
      for (int kt = 0; kt < 6; ++kt)
        at = wmma32(load_af(sA1t, A1S, kt * 32, lane),
                    load_bp(pw, OFF_T1, 8, kt, wv, lane), at);
#pragma unroll
      for (int j = 0; j < 8; ++j) {
        int row   = j + hs * 8;
        float h1n = (1.f - z1v[j]) * hdv[j] + z1v[j] * tanhf(at[j]);
        h1v[j] = h1n;
        _Float16 hh = (_Float16)h1n;
        sA2 [row * A2S + col] = hh;
        sA2t[row * A2S + col] = hh;
      }
      v8f ag;
#pragma unroll
      for (int j = 0; j < 8; ++j) ag[j] = bgh2c;
      ag = wmma32(load_af(sAd, 32, 0, lane),
                  load_bp(pw, OFF_GH2, 8, 0, wv, lane), ag);
#pragma unroll
      for (int j = 0; j < 8; ++j) {
        int row  = j + hs * 8;
        float g  = __expf(-fmaxf(ag[j], 0.f));
        float hd = g * h2v[j];
        hd2v[j] = hd;
        sA2[row * A2S + A2H2 + col] = (_Float16)hd;
      }
    }
    __syncthreads();

    // 5. z2, r2 over A2 (K = 288, 9 chunks); write r2*h2d into A2t
    float z2v[8];
    {
      v8f az, ar;
#pragma unroll
      for (int j = 0; j < 8; ++j) { az[j] = bz2c; ar[j] = br2c; }
#pragma unroll
      for (int kt = 0; kt < 9; ++kt) {
        v16h a = load_af(sA2, A2S, kt * 32, lane);
        az = wmma32(a, load_bp(pw, OFF_Z2, 8, kt, wv, lane), az);
        ar = wmma32(a, load_bp(pw, OFF_R2, 8, kt, wv, lane), ar);
      }
#pragma unroll
      for (int j = 0; j < 8; ++j) {
        int row = j + hs * 8;
        z2v[j] = sigmoidf_(az[j]);
        sA2t[row * A2S + A2H2 + col] =
            (_Float16)(sigmoidf_(ar[j]) * hd2v[j]);
      }
    }
    __syncthreads();

    // 6. t2 over A2t -> h2n (regs + sAh)
    {
      v8f at;
#pragma unroll
      for (int j = 0; j < 8; ++j) at[j] = bt2c;
#pragma unroll
      for (int kt = 0; kt < 9; ++kt)
        at = wmma32(load_af(sA2t, A2S, kt * 32, lane),
                    load_bp(pw, OFF_T2, 8, kt, wv, lane), at);
#pragma unroll
      for (int j = 0; j < 8; ++j) {
        int row   = j + hs * 8;
        float h2n = (1.f - z2v[j]) * hd2v[j] + z2v[j] * tanhf(at[j]);
        h2v[j] = h2n;
        sAh[row * HP_ + col] = (_Float16)h2n;          // pad cols write 0
      }
    }
    __syncthreads();

    // 7. fc1 via WMMA (3 N-tiles cover H3=40 padded to 48)
    if (wv < 3) {
      v8f a;
#pragma unroll
      for (int j = 0; j < 8; ++j) a[j] = bf1c;
#pragma unroll
      for (int kt = 0; kt < 4; ++kt)
        a = wmma32(load_af(sAh, HP_, kt * 32, lane),
                   load_bp(pw, OFF_F1, 3, kt, wv, lane), a);
#pragma unroll
      for (int j = 0; j < 8; ++j)
        sFC1[(j + hs * 8) * 48 + col] = lrelu_(a[j]);
    }
    __syncthreads();

    // 8-10. tiny fc tail on VALU
    for (int i = tid; i < 16 * H9_; i += 256) {
      int r = i / H9_, o = i % H9_;
      float acc = sBf2[o];
#pragma unroll
      for (int k = 0; k < H3_; ++k) acc += sWf2[o * H3_ + k] * sFC1[r * 48 + k];
      sFC2[r * 16 + o] = lrelu_(acc);
    }
    __syncthreads();
    for (int i = tid; i < 16 * H27_; i += 256) {
      int r = i / H27_, o = i % H27_;
      float acc = sBf3[o];
#pragma unroll
      for (int k = 0; k < H9_; ++k) acc += sWf3[o * H9_ + k] * sFC2[r * 16 + k];
      sFC3[r * 8 + o] = lrelu_(acc);
    }
    __syncthreads();
    for (int i = tid; i < 16 * OUT_; i += 256) {
      int r = i / OUT_, o = i % OUT_;
      float acc = sBf4[o];
#pragma unroll
      for (int k = 0; k < H27_; ++k) acc += sWf4[o * H27_ + k] * sFC3[r * 8 + k];
      out_tr[((long)(r0 + r) * T_ + t) * OUT_ + o] = acc;
      if (t == T_ - 1) out_main[(r0 + r) * OUT_ + o] = acc;  // fc_op(h2_final)
    }
    __syncthreads();
  }

  // ------------------------------ epilogue --------------------------------
  // final h2 straight from registers
#pragma unroll
  for (int j = 0; j < 8; ++j) {
    int row = j + hs * 8;
    if (col < H_) out_h2[(long)(r0 + row) * H_ + col] = h2v[j];
  }

  // aux head: a1 via WMMA on final h2 (sAh holds it in f16)
  if (wv < 3) {
    v8f a;
    float bf = (col < H3_) ? p.b_a1[col] : 0.f;
#pragma unroll
    for (int j = 0; j < 8; ++j) a[j] = bf;
#pragma unroll
    for (int kt = 0; kt < 4; ++kt)
      a = wmma32(load_af(sAh, HP_, kt * 32, lane),
                 load_bp(pw, OFF_A1W, 3, kt, wv, lane), a);
#pragma unroll
    for (int j = 0; j < 8; ++j)
      sFC1[(j + hs * 8) * 48 + col] = lrelu_(a[j]);
  }
  __syncthreads();
  for (int i = tid; i < 16 * H9_; i += 256) {
    int r = i / H9_, o = i % H9_;
    float acc = p.b_a2[o];
    for (int k = 0; k < H3_; ++k) acc += p.W_a2[o * H3_ + k] * sFC1[r * 48 + k];
    sFC2[r * 16 + o] = lrelu_(acc);
  }
  __syncthreads();
  for (int i = tid; i < 16 * H27_; i += 256) {
    int r = i / H27_, o = i % H27_;
    float acc = p.b_a3[o];
    for (int k = 0; k < H9_; ++k) acc += p.W_a3[o * H9_ + k] * sFC2[r * 16 + k];
    sFC3[r * 8 + o] = lrelu_(acc);
  }
  __syncthreads();
  for (int i = tid; i < 16 * AUX_; i += 256) {
    int r = i / AUX_, o = i % AUX_;
    float acc = p.b_a4[o];
    for (int k = 0; k < H27_; ++k) acc += p.W_a4[o * H27_ + k] * sFC3[r * 8 + k];
    out_aux[(r0 + r) * AUX_ + o] = acc;
  }
}

// ---------------------------------------------------------------------------

extern "C" void kernel_launch(void* const* d_in, const int* in_sizes, int n_in,
                              void* d_out, int out_size, void* d_ws,
                              size_t ws_size, hipStream_t stream) {
  (void)in_sizes; (void)n_in; (void)out_size; (void)ws_size;
  const float* inputs = (const float*)d_in[0];
  const float* x_mean = (const float*)d_in[3];
  const float* W_gx  = (const float*)d_in[4];  const float* b_gx  = (const float*)d_in[5];
  const float* W_gh1 = (const float*)d_in[6];  const float* b_gh1 = (const float*)d_in[7];
  const float* W_r1  = (const float*)d_in[8];  const float* b_r1  = (const float*)d_in[9];
  const float* W_z1  = (const float*)d_in[10]; const float* b_z1  = (const float*)d_in[11];
  const float* W_t1  = (const float*)d_in[12]; const float* b_t1  = (const float*)d_in[13];
  const float* W_gh2 = (const float*)d_in[14]; const float* b_gh2 = (const float*)d_in[15];
  const float* W_r2  = (const float*)d_in[16]; const float* b_r2  = (const float*)d_in[17];
  const float* W_z2  = (const float*)d_in[18]; const float* b_z2  = (const float*)d_in[19];
  const float* W_t2  = (const float*)d_in[20]; const float* b_t2  = (const float*)d_in[21];
  const float* W_f1  = (const float*)d_in[22]; const float* b_f1  = (const float*)d_in[23];
  const float* W_f2  = (const float*)d_in[24]; const float* b_f2  = (const float*)d_in[25];
  const float* W_f3  = (const float*)d_in[26]; const float* b_f3  = (const float*)d_in[27];
  const float* W_f4  = (const float*)d_in[28]; const float* b_f4  = (const float*)d_in[29];
  const float* W_a1  = (const float*)d_in[30]; const float* b_a1  = (const float*)d_in[31];
  const float* W_a2  = (const float*)d_in[32]; const float* b_a2  = (const float*)d_in[33];
  const float* W_a3  = (const float*)d_in[34]; const float* b_a3  = (const float*)d_in[35];
  const float* W_a4  = (const float*)d_in[36]; const float* b_a4  = (const float*)d_in[37];

  _Float16* pw = (_Float16*)d_ws;
  pack_weights_kernel<<<64, 256, 0, stream>>>(W_gx, W_gh1, W_r1, W_z1, W_t1,
                                              W_gh2, W_r2, W_z2, W_t2, W_f1,
                                              W_a1, pw);

  Params P;
  P.inputs = inputs; P.x_mean = x_mean;
  P.b_gx = b_gx; P.b_gh1 = b_gh1;
  P.b_r1 = b_r1; P.b_z1 = b_z1; P.b_t1 = b_t1;
  P.b_gh2 = b_gh2; P.b_r2 = b_r2; P.b_z2 = b_z2; P.b_t2 = b_t2;
  P.b_f1 = b_f1; P.W_f2 = W_f2; P.b_f2 = b_f2;
  P.W_f3 = W_f3; P.b_f3 = b_f3; P.W_f4 = W_f4; P.b_f4 = b_f4;
  P.b_a1 = b_a1; P.W_a2 = W_a2; P.b_a2 = b_a2;
  P.W_a3 = W_a3; P.b_a3 = b_a3; P.W_a4 = W_a4; P.b_a4 = b_a4;
  P.pw = pw; P.out = (float*)d_out;

  grud_main<<<16, 256, 0, stream>>>(P);
}
